// DiffAttention_23493471109777
// MI455X (gfx1250) — matrix operand order repaired and checked by reading, not compiled, not run on hardware
//
#include <hip/hip_runtime.h>
#include <hip/hip_bf16.h>

// ---------------------------------------------------------------------------
// DiffAttention on gfx1250 (MI455X): WMMA f16->f32 flash-style pipeline.
// B=2, S=2048, H=768, NH=12, HD=32 (2*NH component heads), lambda_init=0.2
// ---------------------------------------------------------------------------

typedef __attribute__((ext_vector_type(16))) _Float16 v16h;
typedef __attribute__((ext_vector_type(8)))  _Float16 v8h;
typedef __attribute__((ext_vector_type(8)))  float    v8f;

constexpr int Bb  = 2;
constexpr int Ss  = 2048;
constexpr int Hh  = 768;
constexpr int NHh = 12;
constexpr int HDd = 32;     // per-component head dim
constexpr float LAMBDA_INIT = 0.2f;
constexpr float Q_SCALE = 0.17677669529663687f;  // 32^-0.5

// ---- WMMA operand loaders (layouts per CDNA5 ISA 7.12.2, wave32) ----------

// A operand: 16x32 f16 (M=rows, K=dims), row-major source with leading dim ld.
// lane m=lane&15, hh=lane>>4; comps 0..7 -> K=hh*8+i, comps 8..15 -> K=16+hh*8+i
__device__ __forceinline__ v16h load_tile_A(const _Float16* p, int ld, int lane) {
  const int m = lane & 15, hh = lane >> 4;
  const _Float16* r = p + m * ld + hh * 8;
  v8h lo = *(const v8h*)(r);
  v8h hi = *(const v8h*)(r + 16);
  v16h a;
#pragma unroll
  for (int i = 0; i < 8; ++i) { a[i] = lo[i]; a[i + 8] = hi[i]; }
  return a;
}

// B operand: 32x16 f16 (K x N). Element (k,n) taken from storage[n][k]
// (row = n, K contiguous), leading dim ld. lane n=lane&15, hh=lane>>4;
// comps 0..7 -> K = hh*8 + c, comps 8..15 -> K = 16 + hh*8 + (c-8).
__device__ __forceinline__ v16h load_tile_B(const _Float16* p, int ld, int lane) {
  const int n = lane & 15, hh = lane >> 4;
  const _Float16* r = p + n * ld + hh * 8;
  v8h lo = *(const v8h*)(r);
  v8h hi = *(const v8h*)(r + 16);
  v16h b;
#pragma unroll
  for (int i = 0; i < 8; ++i) { b[i] = lo[i]; b[i + 8] = hi[i]; }
  return b;
}

__device__ __forceinline__ v8f wmma16(v16h a, v16h b, v8f c) {
  return __builtin_amdgcn_wmma_f32_16x16x32_f16(false, a, false, b, (short)0, c,
                                                false, false);
}

// ---------------------------------------------------------------------------
// Kernel: f32 -> f16 convert
// ---------------------------------------------------------------------------
__global__ void cvt_f32_to_f16(const float* __restrict__ in,
                               _Float16* __restrict__ out, int n) {
  int i = blockIdx.x * blockDim.x + threadIdx.x;
  if (i < n) out[i] = (_Float16)in[i];
}

// ---------------------------------------------------------------------------
// Kernel: W[k][n] f32 -> Wt[n][k] f16 (768x768)
// ---------------------------------------------------------------------------
__global__ void transpose_w(const float* __restrict__ W,
                            _Float16* __restrict__ Wt) {
  int tid = blockIdx.x * blockDim.x + threadIdx.x;
  if (tid >= Hh * Hh) return;
  int n = tid / Hh, k = tid - n * Hh;
  Wt[n * Hh + k] = (_Float16)W[k * Hh + n];
}

// ---------------------------------------------------------------------------
// GEMM: Y[4096x768] = A[4096x768] @ W (via Wt[n][k]) + bias
//   mode 0: f16 out, y = (acc+bias)*scale          (Q / K projections)
//   mode 1: f16 transposed out Vt[b][n][s]         (V projection)
//   mode 2: f32 out, y = acc+bias                  (output projection)
// wave = 16x64 tile; block = 4 waves.
// ---------------------------------------------------------------------------
__global__ void __launch_bounds__(128)
gemm768(const _Float16* __restrict__ A, const _Float16* __restrict__ Bt,
        const float* __restrict__ bias, _Float16* __restrict__ outH,
        float* __restrict__ outF, int mode, float scale) {
  const int lane = threadIdx.x;
  const int mt = blockIdx.y * 4 + threadIdx.y;
  const int m0 = mt * 16;
  const int n0 = blockIdx.x * 64;

  v8f acc[4] = {};
#pragma unroll 4
  for (int k0 = 0; k0 < Hh; k0 += 32) {
    v16h a = load_tile_A(A + (size_t)m0 * Hh + k0, Hh, lane);
#pragma unroll
    for (int t = 0; t < 4; ++t) {
      v16h b = load_tile_B(Bt + (size_t)(n0 + t * 16) * Hh + k0, Hh, lane);
      acc[t] = wmma16(a, b, acc[t]);
    }
  }

  const int col = lane & 15, half = lane >> 4;
  if (mode == 1) {
    // transposed f16 store: Vt[b][n][s], 8 consecutive s per lane
    const int bidx = m0 >> 11;                 // / 2048
    const int s0 = (m0 & 2047) + half * 8;
#pragma unroll
    for (int t = 0; t < 4; ++t) {
      const int n = n0 + t * 16 + col;
      const float bi = bias[n];
      v8h pack;
#pragma unroll
      for (int r = 0; r < 8; ++r) pack[r] = (_Float16)(acc[t][r] + bi);
      *(v8h*)(outH + ((size_t)bidx * Hh + n) * Ss + s0) = pack;
    }
  } else {
#pragma unroll
    for (int t = 0; t < 4; ++t) {
      const int n = n0 + t * 16 + col;
      const float bi = bias[n];
#pragma unroll
      for (int r = 0; r < 8; ++r) {
        const int m = m0 + half * 8 + r;
        const float y = (acc[t][r] + bi) * scale;
        if (mode == 0) outH[(size_t)m * Hh + n] = (_Float16)y;
        else           outF[(size_t)m * Hh + n] = y;
      }
    }
  }
}

// ---------------------------------------------------------------------------
// Differential flash attention.
// One wave per 16-query tile per (b, head). Online softmax for both
// component heads; 32-key blocks; P staged through LDS (f16) for the
// P@V A-operand; V read from transposed Vt[b][dim][s].
// ---------------------------------------------------------------------------
__device__ __forceinline__ void softmax_update(v8f& sa, v8f& sb,
                                               float (&mr)[8], float (&ls)[8],
                                               v8f (&acc)[4],
                                               _Float16* ldsP, int lane) {
  const int col = lane & 15, half = lane >> 4;
#pragma unroll
  for (int r = 0; r < 8; ++r) {
    float v = fmaxf(sa[r], sb[r]);
    v = fmaxf(v, __shfl_xor(v, 1, 32));
    v = fmaxf(v, __shfl_xor(v, 2, 32));
    v = fmaxf(v, __shfl_xor(v, 4, 32));
    v = fmaxf(v, __shfl_xor(v, 8, 32));
    const float mnew = fmaxf(mr[r], v);
    const float fac = __expf(mr[r] - mnew);
    mr[r] = mnew;
    const float pa = __expf(sa[r] - mnew);
    const float pb = __expf(sb[r] - mnew);
    float rs = pa + pb;
    rs += __shfl_xor(rs, 1, 32);
    rs += __shfl_xor(rs, 2, 32);
    rs += __shfl_xor(rs, 4, 32);
    rs += __shfl_xor(rs, 8, 32);
    ls[r] = ls[r] * fac + rs;
#pragma unroll
    for (int t = 0; t < 4; ++t) acc[t][r] *= fac;
    const int row = half * 8 + r;
    ldsP[row * 32 + col]      = (_Float16)pa;
    ldsP[row * 32 + 16 + col] = (_Float16)pb;
  }
}

__global__ void __launch_bounds__(128)
diff_attn(const _Float16* __restrict__ q16, const _Float16* __restrict__ k16,
          const _Float16* __restrict__ vt16,
          const float* __restrict__ lq1, const float* __restrict__ lk1,
          const float* __restrict__ lq2, const float* __restrict__ lk2,
          const float* __restrict__ subw, _Float16* __restrict__ ctx16) {
  __shared__ __align__(16) _Float16 ldsP[4][2][16 * 32];

  const int lane = threadIdx.x;
  const int wv = threadIdx.y;
  const int b = blockIdx.z;
  const int h = blockIdx.y;
  const int q0 = (blockIdx.x * 4 + wv) * 16;

  const _Float16* qp = q16 + ((size_t)(b * Ss + q0)) * Hh + (2 * h) * HDd;
  const v16h aQ1 = load_tile_A(qp, Hh, lane);
  const v16h aQ2 = load_tile_A(qp + HDd, Hh, lane);

  float mr1[8], ls1[8], mr2[8], ls2[8];
#pragma unroll
  for (int r = 0; r < 8; ++r) {
    mr1[r] = -1e30f; mr2[r] = -1e30f; ls1[r] = 0.f; ls2[r] = 0.f;
  }
  v8f acc1[4] = {};
  v8f acc2[4] = {};
  v8f zf = {};

  for (int jb = 0; jb < Ss / 32; ++jb) {
    const int j0 = jb * 32;
    const _Float16* kp = k16 + ((size_t)(b * Ss + j0)) * Hh + (2 * h) * HDd;
    const v16h bK1a = load_tile_B(kp, Hh, lane);
    const v16h bK1b = load_tile_B(kp + 16 * Hh, Hh, lane);
    const v16h bK2a = load_tile_B(kp + HDd, Hh, lane);
    const v16h bK2b = load_tile_B(kp + HDd + 16 * Hh, Hh, lane);

    v8f s1a = wmma16(aQ1, bK1a, zf);
    v8f s1b = wmma16(aQ1, bK1b, zf);
    v8f s2a = wmma16(aQ2, bK2a, zf);
    v8f s2b = wmma16(aQ2, bK2b, zf);

    softmax_update(s1a, s1b, mr1, ls1, acc1, &ldsP[wv][0][0], lane);
    softmax_update(s2a, s2b, mr2, ls2, acc2, &ldsP[wv][1][0], lane);

    __syncthreads();
    asm volatile("s_wait_dscnt 0" ::: "memory");

    const v16h aP1 = load_tile_A(&ldsP[wv][0][0], 32, lane);
    const v16h aP2 = load_tile_A(&ldsP[wv][1][0], 32, lane);
#pragma unroll
    for (int t = 0; t < 4; ++t) {
      const _Float16* vp =
          vt16 + ((size_t)b * Hh + h * 64 + t * 16) * Ss + j0;
      const v16h bV = load_tile_B(vp, Ss, lane);
      acc1[t] = wmma16(aP1, bV, acc1[t]);
      acc2[t] = wmma16(aP2, bV, acc2[t]);
    }
    __syncthreads();
  }

  // lambda_full = exp(lq1.lk1) - exp(lq2.lk2) + lambda_init  (HD=32 = 1 lane each)
  float d1 = lq1[lane] * lk1[lane];
  float d2 = lq2[lane] * lk2[lane];
#pragma unroll
  for (int mk = 1; mk < 32; mk <<= 1) {
    d1 += __shfl_xor(d1, mk, 32);
    d2 += __shfl_xor(d2, mk, 32);
  }
  const float lambda_full = __expf(d1) - __expf(d2) + LAMBDA_INIT;

  const int col = lane & 15, half = lane >> 4;
  float ctx[4][8];
  float ssq[8];
#pragma unroll
  for (int r = 0; r < 8; ++r) ssq[r] = 0.f;
#pragma unroll
  for (int t = 0; t < 4; ++t) {
#pragma unroll
    for (int r = 0; r < 8; ++r) {
      const float c =
          acc1[t][r] / ls1[r] - lambda_full * (acc2[t][r] / ls2[r]);
      ctx[t][r] = c;
      ssq[r] += c * c;
    }
  }
  float rmsf[8];
#pragma unroll
  for (int r = 0; r < 8; ++r) {
    float s = ssq[r];
    s += __shfl_xor(s, 1, 32);
    s += __shfl_xor(s, 2, 32);
    s += __shfl_xor(s, 4, 32);
    s += __shfl_xor(s, 8, 32);
    rmsf[r] = rsqrtf(s * (1.0f / 64.0f) + 1e-5f);
  }
#pragma unroll
  for (int t = 0; t < 4; ++t) {
    const float w = subw[t * 16 + col];
#pragma unroll
    for (int r = 0; r < 8; ++r) {
      const float y = ctx[t][r] * rmsf[r] * w * (1.0f - LAMBDA_INIT);
      const int srow = q0 + half * 8 + r;
      ctx16[((size_t)(b * Ss + srow)) * Hh + h * 64 + t * 16 + col] =
          (_Float16)y;
    }
  }
}

// ---------------------------------------------------------------------------
// Host launcher
// ---------------------------------------------------------------------------
extern "C" void kernel_launch(void* const* d_in, const int* in_sizes, int n_in,
                              void* d_out, int out_size, void* d_ws,
                              size_t ws_size, hipStream_t stream) {
  const float* hidden = (const float*)d_in[0];
  const float* Wq = (const float*)d_in[1];
  const float* bq = (const float*)d_in[2];
  const float* Wk = (const float*)d_in[3];
  const float* bk = (const float*)d_in[4];
  const float* Wv = (const float*)d_in[5];
  const float* bv = (const float*)d_in[6];
  const float* Wo = (const float*)d_in[7];
  const float* bo = (const float*)d_in[8];
  const float* lq1 = (const float*)d_in[9];
  const float* lk1 = (const float*)d_in[10];
  const float* lq2 = (const float*)d_in[11];
  const float* lk2 = (const float*)d_in[12];
  const float* subw = (const float*)d_in[13];
  float* out = (float*)d_out;

  const size_t NTOK = (size_t)Bb * Ss;           // 4096
  const size_t XSZ = NTOK * Hh;                  // 3,145,728
  const size_t WSZ = (size_t)Hh * Hh;            // 589,824

  _Float16* ws = (_Float16*)d_ws;
  _Float16* X16 = ws;
  _Float16* WqT = X16 + XSZ;
  _Float16* WkT = WqT + WSZ;
  _Float16* WvT = WkT + WSZ;
  _Float16* WoT = WvT + WSZ;
  _Float16* Q16 = WoT + WSZ;
  _Float16* K16 = Q16 + XSZ;
  _Float16* VT  = K16 + XSZ;
  _Float16* CTX = VT + XSZ;

  // 1. convert hidden states to f16
  cvt_f32_to_f16<<<(XSZ + 255) / 256, 256, 0, stream>>>(hidden, X16, (int)XSZ);

  // 2. transpose + convert weights
  transpose_w<<<(WSZ + 255) / 256, 256, 0, stream>>>(Wq, WqT);
  transpose_w<<<(WSZ + 255) / 256, 256, 0, stream>>>(Wk, WkT);
  transpose_w<<<(WSZ + 255) / 256, 256, 0, stream>>>(Wv, WvT);
  transpose_w<<<(WSZ + 255) / 256, 256, 0, stream>>>(Wo, WoT);

  // 3. QKV projections (WMMA). grid: (N/64, M/64) with 4 waves/block.
  dim3 gGemm(Hh / 64, (unsigned)(NTOK / 64));
  dim3 bGemm(32, 4);
  gemm768<<<gGemm, bGemm, 0, stream>>>(X16, WqT, bq, Q16, nullptr, 0, Q_SCALE);
  gemm768<<<gGemm, bGemm, 0, stream>>>(X16, WkT, bk, K16, nullptr, 0, 1.0f);
  gemm768<<<gGemm, bGemm, 0, stream>>>(X16, WvT, bv, VT, nullptr, 1, 1.0f);

  // 4. differential flash attention
  dim3 gAttn(Ss / 16 / 4, NHh, Bb);
  dim3 bAttn(32, 4);
  diff_attn<<<gAttn, bAttn, 0, stream>>>(Q16, K16, VT, lq1, lk1, lq2, lk2,
                                         subw, CTX);

  // 5. output projection -> f32 out
  gemm768<<<gGemm, bGemm, 0, stream>>>(CTX, WoT, bo, nullptr, out, 2, 1.0f);
}
